// Blur_498216206770
// MI455X (gfx1250) — matrix operand order repaired and checked by reading, not compiled, not run on hardware
//
#include <hip/hip_runtime.h>
#include <hip/hip_bf16.h>

// -----------------------------------------------------------------------------
// NeRF-style "blur" MLP on MI455X (gfx1250), wave32 + WMMA f16->f32.
//
// ~240 GFLOP vs ~25 MB HBM traffic -> compute bound; all matmuls via
// v_wmma_f32_16x16x32_f16. Register-blocked: each wave holds 16 rows x 256
// cols with all 16 n-tile f32 accumulators (128 VGPRs) live across the k-loop.
//
// Weights: prep kernel packs f32 -> f16 WMMA B-fragments in d_ws
//   fragment = [32 lanes][16 halfs]:
//     lane<16 : col = lane,    halfs j -> K {0..7,16..23}
//     lane>=16: col = lane-16, halfs j -> K {8..15,24..31}
//   K padded to 32; layer 4 (skip concat, K=266) split: k-tile 0 = W4 rows
//   0..9 (input part, padded) + 8 tiles of rows 10..265.
//
// Per k-tile the workgroup stages its 16 B-fragments (16 KB) into LDS with
// GLOBAL_LOAD_ASYNC_TO_LDS_B128 (double buffered, ASYNCcnt + barrier), so the
// DMA of tile kt+1 overlaps the 16 WMMAs of tile kt and global weight traffic
// is once per workgroup instead of once per wave.
//
// A fragment (16x32 f16): lane<16: row=lane,  K {0..7,16..23}
//                         lane>=16: row=lane-16, K {8..15,24..31}
// C/D 16x16 f32: VGPR r, lane L -> (M = r + 8*(L>=16), N = L&15)
// -----------------------------------------------------------------------------

typedef _Float16 v16h __attribute__((ext_vector_type(16)));
typedef _Float16 h8   __attribute__((ext_vector_type(8)));
typedef float    v8f  __attribute__((ext_vector_type(8)));

union ABFrag { v16h v; h8 h[2]; };

#define TILE_M 64
#define HSTR   264   // 256 + 8 halfs pad: row stride 528B -> conflict-free rotation

struct Params {
    const _Float16* frags[8];
    const float*    bias[8];
};

// ---------------------------------------------------------------- prep kernel
__global__ void pack_weights(const float* __restrict__ W, _Float16* __restrict__ dst,
                             int Kin, int Nout, int nkt, int ntl, int split_first) {
    int tid = blockIdx.x * blockDim.x + threadIdx.x;
    int total = nkt * ntl * 512;
    if (tid >= total) return;
    int frag   = tid >> 9;
    int within = tid & 511;
    int lane   = within >> 4;
    int j      = within & 15;
    int kt = frag / ntl;
    int nt = frag - kt * ntl;
    int n  = nt * 16 + (lane & 15);
    int kk = ((lane >> 4) << 3) + (j & 7) + ((j >> 3) << 4);  // 0..31
    int ksrc; bool valid;
    if (split_first) {                         // layer 4: tile 0 = concat input rows
        if (kt == 0) { ksrc = kk;                      valid = (kk < 10);    }
        else         { ksrc = 10 + (kt - 1) * 32 + kk; valid = (ksrc < Kin); }
    } else           { ksrc = kt * 32 + kk;            valid = (ksrc < Kin); }
    valid = valid && (n < Nout);
    float v = valid ? W[(size_t)ksrc * Nout + n] : 0.0f;
    dst[tid] = (_Float16)v;
}

// ---------------------------------------------------------------- main kernel
__global__ __launch_bounds__(128) void blur_mlp(
    const float* __restrict__ dist, const float* __restrict__ opac,
    const int* __restrict__ idx, float* __restrict__ out, Params p, int N)
{
    __shared__ __align__(16) _Float16 inpT[TILE_M][32];        //  4 KB
    __shared__ __align__(16) _Float16 hbuf[2][TILE_M][HSTR];   // 66 KB
    __shared__ __align__(16) _Float16 bbuf[2][16 * 512];       // 32 KB (B-tile dbl buffer)
    __shared__ __align__(16) float    smax[TILE_M][8];         //  2 KB

    const int t    = threadIdx.x;
    const int lane = t & 31;
    const int wave = t >> 5;
    const int mr   = wave << 4;                 // this wave's 16 rows
    const long long base = (long long)blockIdx.x * TILE_M;

    // wave-relative LDS byte addresses (flat LDS aperture: addr[31:0] = LDS offset)
    const unsigned bboff0 = (unsigned)(unsigned long long)(void*)&bbuf[0][0];
    const unsigned bboff1 = (unsigned)(unsigned long long)(void*)&bbuf[1][0];

    // ---- stage padded input tile: cols 0..4 = distances, 5..9 = gathered densities
    {
        int r  = t >> 1;
        int c0 = (t & 1) << 4;
        long long g = base + r;
        #pragma unroll
        for (int c = 0; c < 16; ++c) {
            int col = c0 + c;
            float v = 0.0f;
            if (g < N) {
                if (col < 5)       v = dist[g * 5 + col];
                else if (col < 10) v = opac[idx[g * 5 + (col - 5)]];
            }
            inpT[r][col] = (_Float16)v;
        }
    }
    __syncthreads();

    const int row = lane & 15;
    const int off = (lane >> 4) << 3;           // 0 or 8 halfs

    auto load_a = [&](const _Float16* rb, int kb) -> v16h {
        ABFrag a;
        a.h[0] = *(const h8*)(rb + kb + off);
        a.h[1] = *(const h8*)(rb + kb + 16 + off);
        return a.v;
    };

    // async DMA of one 16-fragment B tile (16 KB) into LDS: 8 x b128 per thread
    auto stage16 = [&](const _Float16* src, unsigned ldsbyte) {
        unsigned long long sbase = (unsigned long long)(const void*)src;
        #pragma unroll
        for (int i = 0; i < 8; ++i) {
            unsigned vo = (unsigned)(i * 2048) + ((unsigned)t << 4);
            asm volatile("global_load_async_to_lds_b128 %0, %1, %2"
                         :: "v"(ldsbyte + vo), "v"(vo), "s"(sbase) : "memory");
        }
    };
    auto wait_async = [&]() {
        asm volatile("s_wait_asynccnt 0x0" ::: "memory");
    };

    // hidden layer (Nout=256, 16 n-tiles), B tiles staged through LDS
    auto do_layer16 = [&](const _Float16* frags, const float* bias,
                          const _Float16 (*hin)[HSTR], _Float16 (*hout)[HSTR],
                          int nktot, bool use_inp) {
        v8f acc[16];
        #pragma unroll
        for (int nt = 0; nt < 16; ++nt) acc[nt] = (v8f){};

        stage16(frags, bboff0);                 // preload tile 0
        wait_async();
        __syncthreads();

        for (int ktl = 0; ktl < nktot; ++ktl) {
            if (ktl + 1 < nktot)                // overlap DMA of next tile
                stage16(frags + (size_t)(ktl + 1) * 16 * 512,
                        ((ktl + 1) & 1) ? bboff1 : bboff0);

            const _Float16* arow;
            int kb;
            if (use_inp && ktl == 0) { arow = &inpT[mr + row][0];      kb = 0; }
            else if (use_inp)        { arow = &hin[mr + row][0];       kb = (ktl - 1) * 32; }
            else                     { arow = &hin[mr + row][0];       kb = ktl * 32; }
            v16h a = load_a(arow, kb);

            const _Float16* bb = &bbuf[ktl & 1][0];
            #pragma unroll
            for (int nt = 0; nt < 16; ++nt) {
                ABFrag b;
                const h8* q = (const h8*)(bb + nt * 512 + (lane << 4));
                b.h[0] = q[0];
                b.h[1] = q[1];
                acc[nt] = __builtin_amdgcn_wmma_f32_16x16x32_f16(
                              false, a, false, b.v, (short)0, acc[nt], false, false);
            }
            wait_async();                       // own DMA done before barrier
            __syncthreads();                    // all waves done with this tile
        }

        const int n  = lane & 15;
        const int m0 = mr + ((lane >> 4) << 3);
        #pragma unroll
        for (int nt = 0; nt < 16; ++nt) {
            const int col = nt * 16 + n;
            float bv = bias[col];
            #pragma unroll
            for (int r8 = 0; r8 < 8; ++r8) {
                float v = acc[nt][r8] + bv;
                hout[m0 + r8][col] = (_Float16)(v > 0.0f ? v : 0.0f);
            }
        }
    };

    // final layer (256 -> 5): one n-tile, 8 k-tiles, B direct from global
    auto do_layer1 = [&](const _Float16* frags, const float* bias,
                         const _Float16 (*hin)[HSTR]) {
        v8f acc = {};
        const _Float16* arow = &hin[mr + row][0];
        for (int kt = 0; kt < 8; ++kt) {
            v16h a = load_a(arow, kt * 32);
            ABFrag b;
            const h8* q = (const h8*)(frags + ((size_t)kt << 9) + (lane << 4));
            b.h[0] = q[0];
            b.h[1] = q[1];
            acc = __builtin_amdgcn_wmma_f32_16x16x32_f16(
                      false, a, false, b.v, (short)0, acc, false, false);
        }
        const int n  = lane & 15;
        const int m0 = mr + ((lane >> 4) << 3);
        if (n < 5) {
            float bv = bias[n];
            #pragma unroll
            for (int r8 = 0; r8 < 8; ++r8)
                smax[m0 + r8][n] = acc[r8] + bv;
        }
    };

    do_layer16(p.frags[0], p.bias[0], nullptr, hbuf[0], 1, true);  __syncthreads();
    do_layer16(p.frags[1], p.bias[1], hbuf[0], hbuf[1], 8, false); __syncthreads();
    do_layer16(p.frags[2], p.bias[2], hbuf[1], hbuf[0], 8, false); __syncthreads();
    do_layer16(p.frags[3], p.bias[3], hbuf[0], hbuf[1], 8, false); __syncthreads();
    do_layer16(p.frags[4], p.bias[4], hbuf[1], hbuf[0], 9, true);  __syncthreads(); // skip concat
    do_layer16(p.frags[5], p.bias[5], hbuf[0], hbuf[1], 8, false); __syncthreads();
    do_layer16(p.frags[6], p.bias[6], hbuf[1], hbuf[0], 8, false); __syncthreads();
    do_layer1 (p.frags[7], p.bias[7], hbuf[0]);                    __syncthreads();

    // ---- softmax over 5 logits, one thread per row
    if (t < TILE_M) {
        long long g = base + t;
        if (g < N) {
            float x[5], mx = -3.4e38f;
            #pragma unroll
            for (int c = 0; c < 5; ++c) { x[c] = smax[t][c]; mx = x[c] > mx ? x[c] : mx; }
            float s = 0.0f;
            #pragma unroll
            for (int c = 0; c < 5; ++c) { x[c] = __expf(x[c] - mx); s += x[c]; }
            float inv = 1.0f / s;
            #pragma unroll
            for (int c = 0; c < 5; ++c) out[g * 5 + c] = x[c] * inv;
        }
    }
}

// indices pass-through (second tuple output)
__global__ void copy_idx(const int* __restrict__ src, int* __restrict__ dst, int n) {
    int i = blockIdx.x * blockDim.x + threadIdx.x;
    if (i < n) dst[i] = src[i];
}

// ---------------------------------------------------------------- launcher
extern "C" void kernel_launch(void* const* d_in, const int* in_sizes, int n_in,
                              void* d_out, int out_size, void* d_ws, size_t ws_size,
                              hipStream_t stream) {
    // d_in order: distances, opacity, indices, W0..W7, b0..b7
    const float* dist = (const float*)d_in[0];
    const float* opac = (const float*)d_in[1];
    const int*   idx  = (const int*)d_in[2];
    const float* Ws[8];
    const float* bs[8];
    for (int l = 0; l < 8; ++l) {
        Ws[l] = (const float*)d_in[3 + l];
        bs[l] = (const float*)d_in[11 + l];
    }
    const int N = in_sizes[0] / 5;

    static const int nkt[8]  = {1, 8, 8, 8, 9, 8, 8, 8};   // k-tiles (incl. padded input tiles)
    static const int ntl[8]  = {16,16,16,16,16,16,16, 1};
    static const int Kin[8]  = {10,256,256,256,266,256,256,256};
    static const int Nout[8] = {256,256,256,256,256,256,256, 5};

    _Float16* wbase = (_Float16*)d_ws;          // needs ~828 KB of d_ws
    size_t offs[8], off = 0;
    for (int l = 0; l < 8; ++l) { offs[l] = off; off += (size_t)nkt[l] * ntl[l] * 512; }

    Params p;
    for (int l = 0; l < 8; ++l) {
        int total = nkt[l] * ntl[l] * 512;
        pack_weights<<<(total + 255) / 256, 256, 0, stream>>>(
            Ws[l], wbase + offs[l], Kin[l], Nout[l], nkt[l], ntl[l], (l == 4) ? 1 : 0);
        p.frags[l] = wbase + offs[l];
        p.bias[l]  = bs[l];
    }

    blur_mlp<<<(N + TILE_M - 1) / TILE_M, 128, 0, stream>>>(
        dist, opac, idx, (float*)d_out, p, N);

    copy_idx<<<(N * 5 + 255) / 256, 256, 0, stream>>>(
        idx, (int*)((float*)d_out + (size_t)N * 5), N * 5);
}
